// MyLSTM_42339787604167
// MI455X (gfx1250) — matrix operand order repaired and checked by reading, not compiled, not run on hardware
//
#include <hip/hip_runtime.h>
#include <hip/hip_bf16.h>
#include <math.h>

#define T_SEQ 512
#define B_SZ  64
#define E_SZ  128
#define H_SZ  128
#define ZK    256          // E + H
#define LDSP  264          // padded weight row length (bf16) -> conflict-free b128 lds loads
#define XP    136          // padded activation row length (bf16), 272B stride, 16B aligned
#define C_SZ  5

typedef __attribute__((ext_vector_type(8)))  float  v8f;
typedef __attribute__((ext_vector_type(16))) __bf16 v16bf;

union FragB16 { v16bf v; uint4 q[2]; unsigned short s[16]; };

static __device__ __forceinline__ unsigned short f32_to_bf16(float f) {
    union { float f; unsigned u; } x; x.f = f;
    unsigned r = x.u + 0x7FFFu + ((x.u >> 16) & 1u);   // round-to-nearest-even
    return (unsigned short)(r >> 16);
}
static __device__ __forceinline__ float sigmoidf_(float x) {
    return 1.0f / (1.0f + __expf(-x));
}
// async copy of 16 bytes global -> LDS (ASYNCcnt-tracked, CDNA5)
static __device__ __forceinline__ void async_g2l_b128(unsigned lds_off,
                                                      const unsigned short* gaddr) {
    asm volatile("global_load_async_to_lds_b128 %0, %1, off"
                 :: "v"(lds_off), "v"(gaddr) : "memory");
}
static __device__ __forceinline__ void wait_asynccnt0() {
    asm volatile("s_wait_asynccnt 0x0" ::: "memory");
}

// ---------------------------------------------------------------------------
// Kernel 0: embedding gather, fp32 -> bf16, layout (T, B, E)
// ---------------------------------------------------------------------------
__global__ void gather_embed_bf16(const int* __restrict__ x,
                                  const float* __restrict__ embed,
                                  unsigned short* __restrict__ xe) {
    int bt = blockIdx.x;                 // bt = b*T + t  (x is (B,T) row-major)
    int b  = bt / T_SEQ;
    int t  = bt % T_SEQ;
    int id = x[bt];
    int e  = threadIdx.x;                // 128 threads = E
    float v = embed[(long)id * E_SZ + e];
    xe[((long)t * B_SZ + b) * E_SZ + e] = f32_to_bf16(v);
}

// ---------------------------------------------------------------------------
// Kernel 1: persistent per-direction LSTM. One workgroup (8 wave32) per dir.
// Weights resident in LDS (bf16) for the whole sequence; x tiles streamed in
// via async global->LDS (double buffered); h lives in its own LDS buffer.
// ---------------------------------------------------------------------------
struct LstmParams {
    const float* W[2][4];     // [dir][f,i,c,o] each (H, E+H) row-major fp32
    const float* bias[2][4];  // [dir][f,i,c,o] each (H,)
    const unsigned short* xe; // (T, B, E) bf16
    float* hmax;              // [2][B][H] fp32 running max of h over t
};

__global__ __launch_bounds__(256)
void lstm_dir_kernel(LstmParams p) {
    __shared__ unsigned short wlds[4][H_SZ][LDSP];   // 264.0 KB weights (bf16)
    __shared__ unsigned short xbuf[2][B_SZ][XP];     //  34.0 KB x tiles, ping-pong
    __shared__ unsigned short hbuf[B_SZ][XP];        //  17.0 KB h state

    const int dir  = blockIdx.x;         // 0 = forward, 1 = backward
    const int tid  = threadIdx.x;
    const int lane = tid & 31;
    const int wave = tid >> 5;           // 0..7 -> owns h columns [16w, 16w+16)
    const int h0   = wave * 16;
    const int nn   = lane & 15;
    const int hiHalf  = (lane >> 4);     // 0: lanes 0-15, 1: lanes 16-31
    const int a_kbase = hiHalf ? 8  : 0; // A frag K sub-base (16-bit A 16x32 layout)
    const int b_kbase = hiHalf ? 8  : 0; // B frag K sub-base (same K order as A)

    // ---- stage 1: weights fp32 -> bf16 into LDS (coalesced) ----
    for (int f = tid; f < 4 * H_SZ * ZK; f += 256) {
        int g = f >> 15;                 // / (128*256)
        int r = (f >> 8) & 127;
        int c = f & 255;
        wlds[g][r][c] = f32_to_bf16(p.W[dir][g][r * ZK + c]);
    }
    // h0 = 0
    for (int f = tid; f < B_SZ * H_SZ; f += 256) {
        hbuf[f >> 7][f & 127] = 0;
    }
    // async-load x(t=0) into xbuf[0]
    {
        long te = dir ? (long)(T_SEQ - 1) : 0;
        const unsigned short* g = p.xe + te * (long)B_SZ * E_SZ;
        #pragma unroll
        for (int i = 0; i < 4; ++i) {
            int elem = (tid + 256 * i) * 8;
            unsigned ldsa = (unsigned)(uintptr_t)&xbuf[0][elem >> 7][elem & 127];
            async_g2l_b128(ldsa, g + elem);
        }
    }

    // per-lane gate biases (bias depends only on output column = N = lane&15)
    float biasr[4];
    #pragma unroll
    for (int g = 0; g < 4; ++g) biasr[g] = p.bias[dir][g][h0 + nn];

    v8f hmaxacc[4];
    #pragma unroll
    for (int mt = 0; mt < 4; ++mt)
        #pragma unroll
        for (int e = 0; e < 8; ++e) hmaxacc[mt][e] = -1e30f;

    wait_asynccnt0();
    __syncthreads();

    for (int t = 0; t < T_SEQ; ++t) {
        // kick off async copy of x_{t+1} into the other x buffer (overlaps WMMA)
        if (t + 1 < T_SEQ) {
            long te = dir ? (long)(T_SEQ - 2 - t) : (long)(t + 1);
            const unsigned short* g = p.xe + te * (long)B_SZ * E_SZ;
            int buf = (t + 1) & 1;
            #pragma unroll
            for (int i = 0; i < 4; ++i) {
                int elem = (tid + 256 * i) * 8;
                unsigned ldsa = (unsigned)(uintptr_t)&xbuf[buf][elem >> 7][elem & 127];
                async_g2l_b128(ldsa, g + elem);
            }
        }
        if (t + 2 < T_SEQ) {
            long te2 = dir ? (long)(T_SEQ - 3 - t) : (long)(t + 2);
            __builtin_prefetch(p.xe + te2 * (long)B_SZ * E_SZ + tid * 8, 0, 1);
        }

        const unsigned short (*xc)[XP] = xbuf[t & 1];

        // kt-outer: 8 fragment loads feed 16 WMMAs per K-tile (one dscnt wait each)
        v8f acc[4][4] = {};                       // [m-tile][gate f,i,cbar,o]
        #pragma unroll
        for (int kt = 0; kt < 8; ++kt) {
            FragB16 af[4], bfr[4];
            #pragma unroll
            for (int mt = 0; mt < 4; ++mt) {      // A frags: K 0-127 from x, 128-255 from h
                const unsigned short* zr = (kt < 4)
                    ? &xc[mt * 16 + nn][kt * 32 + a_kbase]
                    : &hbuf[mt * 16 + nn][(kt - 4) * 32 + a_kbase];
                af[mt].q[0] = *(const uint4*)(zr);
                af[mt].q[1] = *(const uint4*)(zr + 16);
            }
            #pragma unroll
            for (int g = 0; g < 4; ++g) {         // B frags: one per gate
                const unsigned short* wr = &wlds[g][h0 + nn][kt * 32 + b_kbase];
                bfr[g].q[0] = *(const uint4*)(wr);
                bfr[g].q[1] = *(const uint4*)(wr + 16);
            }
            #pragma unroll
            for (int mt = 0; mt < 4; ++mt)
                #pragma unroll
                for (int g = 0; g < 4; ++g)
                    acc[mt][g] = __builtin_amdgcn_wmma_f32_16x16x32_bf16(
                        false, af[mt].v, false, bfr[g].v, (short)0, acc[mt][g],
                        false, false);
        }

        // gate nonlinearities + source-faithful c_new = (f+i)*cbar
        v8f hstage[4];
        #pragma unroll
        for (int mt = 0; mt < 4; ++mt) {
            #pragma unroll
            for (int e = 0; e < 8; ++e) {
                float fg = sigmoidf_(acc[mt][0][e] + biasr[0]);
                float ig = sigmoidf_(acc[mt][1][e] + biasr[1]);
                float cg = tanhf   (acc[mt][2][e] + biasr[2]);
                float og = sigmoidf_(acc[mt][3][e] + biasr[3]);
                float hn = og * tanhf((fg + ig) * cg);
                hstage[mt][e]  = hn;
                hmaxacc[mt][e] = fmaxf(hmaxacc[mt][e], hn);
            }
        }

        __syncthreads();   // everyone done reading hbuf / xbuf[cur] for step t

        // commit h_{t+1} (D layout: lane = N, vgpr = M, +8 on upper half-wave)
        #pragma unroll
        for (int mt = 0; mt < 4; ++mt)
            #pragma unroll
            for (int e = 0; e < 8; ++e) {
                int brow = mt * 16 + e + (hiHalf ? 8 : 0);
                hbuf[brow][h0 + nn] = f32_to_bf16(hstage[mt][e]);
            }

        wait_asynccnt0();  // x_{t+1} fully in LDS (issuing wave's portion)
        __syncthreads();   // ... and visible to all waves
    }

    // write running max (tanh + head applied by kernel 2)
    float* hm = p.hmax + (long)dir * B_SZ * H_SZ;
    #pragma unroll
    for (int mt = 0; mt < 4; ++mt)
        #pragma unroll
        for (int e = 0; e < 8; ++e) {
            int brow = mt * 16 + e + (hiHalf ? 8 : 0);
            hm[brow * H_SZ + h0 + nn] = hmaxacc[mt][e];
        }
}

// ---------------------------------------------------------------------------
// Kernel 2: out[b,c] = sum_j tanh(hmax[concat]) * Wout[c,j] + bout[c]
// ---------------------------------------------------------------------------
__global__ void head_kernel(const float* __restrict__ hmax,
                            const float* __restrict__ Wout,
                            const float* __restrict__ bout,
                            float* __restrict__ out) {
    int tidg = blockIdx.x * blockDim.x + threadIdx.x;
    if (tidg >= B_SZ * C_SZ) return;
    int b = tidg / C_SZ, c = tidg % C_SZ;
    float acc = bout[c];
    for (int j = 0; j < 2 * H_SZ; ++j) {
        float hv = (j < H_SZ) ? hmax[b * H_SZ + j]
                              : hmax[B_SZ * H_SZ + b * H_SZ + (j - H_SZ)];
        acc += tanhf(hv) * Wout[c * 2 * H_SZ + j];
    }
    out[b * C_SZ + c] = acc;
}

// ---------------------------------------------------------------------------
extern "C" void kernel_launch(void* const* d_in, const int* in_sizes, int n_in,
                              void* d_out, int out_size, void* d_ws, size_t ws_size,
                              hipStream_t stream) {
    const int*   x     = (const int*)d_in[0];
    const float* embed = (const float*)d_in[1];

    LstmParams p;
    // d_in[2..9]  = Wf_f,bf_f,Wi_f,bi_f,Wc_f,bc_f,Wo_f,bo_f
    // d_in[10..17]= Wf_b,bf_b,Wi_b,bi_b,Wc_b,bc_b,Wo_b,bo_b
    for (int dir = 0; dir < 2; ++dir)
        for (int g = 0; g < 4; ++g) {
            p.W[dir][g]    = (const float*)d_in[2 + dir * 8 + g * 2];
            p.bias[dir][g] = (const float*)d_in[2 + dir * 8 + g * 2 + 1];
        }
    const float* Wout = (const float*)d_in[18];
    const float* bout = (const float*)d_in[19];

    unsigned short* xe   = (unsigned short*)d_ws;                       // T*B*E bf16 (8.4 MB)
    float*          hmax = (float*)((char*)d_ws +
                                    (size_t)T_SEQ * B_SZ * E_SZ * 2);   // 2*B*H fp32
    p.xe = xe;
    p.hmax = hmax;

    gather_embed_bf16<<<B_SZ * T_SEQ, E_SZ, 0, stream>>>(x, embed, xe);
    lstm_dir_kernel<<<2, 256, 0, stream>>>(p);
    head_kernel<<<(B_SZ * C_SZ + 255) / 256, 256, 0, stream>>>(hmax, Wout, bout,
                                                               (float*)d_out);
}
